// AssembledBlock_69672959476330
// MI455X (gfx1250) — matrix operand order repaired and checked by reading, not compiled, not run on hardware
//
#include <hip/hip_runtime.h>
#include <hip/hip_bf16.h>

typedef __attribute__((ext_vector_type(16))) __bf16          v16bf;
typedef __attribute__((ext_vector_type(16))) unsigned short  v16u;
typedef __attribute__((ext_vector_type(8)))  float           v8f;

#define BS   16
#define CIN  64
#define COUT 64
#define HH   128
#define WW   128
#define KDIM 576            // 9 taps * 64 ci, K' = tap*64 + ci
#define HID  16
#define NE   4
#define PW   130            // padded NHWC width/height
#define PROW (PW * 64)      // elements per padded row  (8320)
#define PSAMP ((size_t)PW * PROW) // elements per padded sample (1081600)
#define TP   66             // transpose LDS pad (bank-conflict mitigation)

__device__ __forceinline__ unsigned short f2bf(float f) {
  unsigned int u = __float_as_uint(f);
  u += 0x7fffu + ((u >> 16) & 1u);   // round-to-nearest-even
  return (unsigned short)(u >> 16);
}

// ---- K1: NCHW f32 -> padded NHWC bf16 (LDS-tiled transpose) + avg-pool partial sums ----
__global__ __launch_bounds__(256) void k_cvt(const float* __restrict__ x,
                                             unsigned short* __restrict__ act,
                                             float* __restrict__ pooled /* sums */) {
  __shared__ unsigned short sT[128 * TP];
  __shared__ float sSum[64];
  const int b = blockIdx.x >> 7;
  const int h = blockIdx.x & 127;
  const int t = threadIdx.x;
  if (t < 64) sSum[t] = 0.f;
  __syncthreads();
  for (int i = t; i < 64 * 32; i += 256) {
    int c = i >> 5, q = i & 31;
    const float4 v = *(const float4*)(x + ((size_t)(b * 64 + c) * HH + h) * WW + q * 4);
    atomicAdd(&sSum[c], v.x + v.y + v.z + v.w);
    unsigned short* dst = &sT[(q * 4) * TP + c];
    dst[0 * TP] = f2bf(v.x);
    dst[1 * TP] = f2bf(v.y);
    dst[2 * TP] = f2bf(v.z);
    dst[3 * TP] = f2bf(v.w);
  }
  __syncthreads();
  if (t < 64) atomicAdd(&pooled[b * 64 + t], sSum[t]);
  unsigned short* orow = act + (size_t)b * PSAMP + (size_t)(h + 1) * PROW + 64;
  for (int j = t; j < 64 * 128 / 8; j += 256) {
    int base = j * 8;
    int w = base >> 6, c0 = base & 63;
    uint4 pk; unsigned int* pw = (unsigned int*)&pk;
    #pragma unroll
    for (int p = 0; p < 4; ++p) {
      unsigned int a0 = sT[w * TP + c0 + 2 * p];
      unsigned int a1 = sT[w * TP + c0 + 2 * p + 1];
      pw[p] = a0 | (a1 << 16);
    }
    *(uint4*)(orow + base) = pk;
  }
}

// ---------------- K2: control MLP + softmax + mixed biases ----------------
__global__ __launch_bounds__(256) void k_ctrl(const float* __restrict__ pooled, // sums
                                              const float* __restrict__ w1,    // [16,64]
                                              const float* __restrict__ w2,    // [256,16]
                                              const float* __restrict__ b1,
                                              const float* __restrict__ b2,
                                              const float* __restrict__ b3,
                                              float* __restrict__ coeff,       // [16,64,4]
                                              float* __restrict__ abias) {     // [3,16,64]
  __shared__ float sP[BS * CIN];
  __shared__ float sH[BS * HID];
  __shared__ float sC[BS * COUT * NE];
  const int t = threadIdx.x;
  for (int i = t; i < BS * CIN; i += 256) sP[i] = pooled[i] * (1.0f / (HH * WW));
  __syncthreads();
  {
    int b = t >> 4, j = t & 15;
    float a = 0.f;
    for (int c = 0; c < CIN; ++c) a += sP[b * CIN + c] * w1[j * CIN + c];
    sH[b * HID + j] = a > 0.f ? a : 0.f;
  }
  __syncthreads();
  for (int g = t; g < BS * COUT; g += 256) {
    int b = g >> 6, c = g & 63;
    float l[NE], m = -1e30f;
    for (int e = 0; e < NE; ++e) {
      float a = 0.f;
      for (int j = 0; j < HID; ++j) a += sH[b * HID + j] * w2[(c * NE + e) * HID + j];
      l[e] = a * (1.0f / 30.0f);
      m = fmaxf(m, l[e]);
    }
    float sum = 0.f;
    for (int e = 0; e < NE; ++e) { l[e] = __expf(l[e] - m); sum += l[e]; }
    float inv = 1.0f / sum;
    for (int e = 0; e < NE; ++e) {
      float v = l[e] * inv;
      sC[g * NE + e] = v;
      coeff[g * NE + e] = v;
    }
  }
  __syncthreads();
  for (int i = t; i < 3 * BS * COUT; i += 256) {
    int layer = i / (BS * COUT);
    int r = i % (BS * COUT);
    int b = r >> 6, c = r & 63;
    const float* bb = layer == 0 ? b1 : (layer == 1 ? b2 : b3);
    float a = 0.f;
    for (int e = 0; e < NE; ++e) a += sC[(b * COUT + c) * NE + e] * bb[e * COUT + c];
    abias[i] = a;
  }
}

// -------- K3: expert-mix weights into bf16 [3][16][64][K'] with K' = tap*64 + ci --------
__global__ __launch_bounds__(256) void k_mixw(const float* __restrict__ coeff,
                                              const float* __restrict__ wt1,
                                              const float* __restrict__ wt2,
                                              const float* __restrict__ wt3,
                                              unsigned short* __restrict__ awbf) {
  const int per = BS * COUT * KDIM;
  int idx = blockIdx.x * 256 + threadIdx.x;
  if (idx >= 3 * per) return;
  int layer = idx / per;
  int r  = idx % per;
  int b  = r / (COUT * KDIM);
  int r2 = r % (COUT * KDIM);
  int o  = r2 / KDIM;
  int kp = r2 % KDIM;                 // K' = tap*64 + ci
  int tap = kp >> 6, ci = kp & 63;
  int knat = ci * 9 + tap;            // source layout: i*9 + kh*3 + kw
  const float* wt = layer == 0 ? wt1 : (layer == 1 ? wt2 : wt3);
  const float* cf = coeff + (b * COUT + o) * NE;
  float a = 0.f;
  for (int e = 0; e < NE; ++e) a += cf[e] * wt[((size_t)e * COUT + o) * KDIM + knat];
  awbf[idx] = f2bf(a);
}

// -------- K4: implicit-GEMM 3x3 conv, v_wmma_f32_16x16x32_bf16, pipelined fragments ----
union AFrag { uint4 q[2]; v16bf v; };
union BFrag { v16u u; v16bf v; };

__device__ __forceinline__ void load_a(AFrag& A, const unsigned short* wrow, int k0, int hf) {
  A.q[0] = *(const uint4*)&wrow[k0 + hf * 16];
  A.q[1] = *(const uint4*)&wrow[k0 + hf * 16 + 8];
}

__device__ __forceinline__ void load_b(BFrag* B, const unsigned short* samp, int h,
                                       int k0, int hf, int nbase, int lrow) {
  const int Kb  = k0 + hf * 16;
  const int tap = Kb >> 6;
  const int cb  = Kb & 63;
  const int kh  = tap / 3, kw = tap % 3;
  const unsigned short* rowp = samp + (size_t)(h + kh) * PROW + (size_t)kw * 64 + cb;
  #pragma unroll
  for (int nt = 0; nt < 4; ++nt) {
    const int wcol = nbase + nt * 16 + lrow;
    B[nt].u = *(const v16u*)(rowp + (size_t)wcol * 64);   // 32B contiguous, 32B aligned
  }
}

__global__ __launch_bounds__(256) void k_conv(const unsigned short* __restrict__ act,  // padded NHWC bf16
                                              const unsigned short* __restrict__ wk,   // bf16 [16][64][576]
                                              const float* __restrict__ bias,          // f32 [16][64]
                                              unsigned short* __restrict__ outh,       // padded NHWC bf16 (L1-2)
                                              float* __restrict__ outf) {              // NCHW f32 (L3)
  extern __shared__ char smem[];
  unsigned short* sW = (unsigned short*)smem;          // 64*576 bf16 = 72KB
  float* sBias = (float*)(sW + COUT * KDIM);

  const int b = blockIdx.x >> 7;
  const int h = blockIdx.x & 127;
  const int t = threadIdx.x;

  {
    const uint4* src = (const uint4*)(wk + (size_t)b * COUT * KDIM);
    uint4* dst = (uint4*)sW;
    for (int i = t; i < COUT * KDIM / 8; i += 256) dst[i] = src[i];
  }
  if (t < COUT) sBias[t] = bias[b * COUT + t];
  __syncthreads();

  const int lane  = t & 31;
  const int wave  = t >> 5;
  const int hf    = lane >> 4;
  const int lrow  = lane & 15;
  const int mtile = wave & 3;
  const int nbase = (wave >> 2) * 64;

  const unsigned short* samp = act + (size_t)b * PSAMP;
  const unsigned short* wrow = sW + (mtile * 16 + lrow) * KDIM;

  v8f acc[4] = {};
  AFrag Af[2];
  BFrag Bf[2][4];

  load_a(Af[0], wrow, 0, hf);
  load_b(Bf[0], samp, h, 0, hf, nbase, lrow);

  #pragma unroll
  for (int ks = 0; ks < KDIM / 32; ++ks) {       // fully unrolled; ping-pong regs SSA-renamed
    const int cur = ks & 1, nxt = cur ^ 1;
    if (ks < KDIM / 32 - 1) {                    // prefetch next K-step while current computes
      load_a(Af[nxt], wrow, (ks + 1) * 32, hf);
      load_b(Bf[nxt], samp, h, (ks + 1) * 32, hf, nbase, lrow);
    }
    #pragma unroll
    for (int nt = 0; nt < 4; ++nt)
      acc[nt] = __builtin_amdgcn_wmma_f32_16x16x32_bf16(false, Af[cur].v, false, Bf[cur][nt].v,
                                                        (short)0, acc[nt], false, false);
  }

  if (outf) {   // final layer: f32 NCHW, coalesced along w
    #pragma unroll
    for (int nt = 0; nt < 4; ++nt) {
      const int wcol = nbase + nt * 16 + lrow;
      #pragma unroll
      for (int r = 0; r < 8; ++r) {
        int o = mtile * 16 + hf * 8 + r;
        outf[(((size_t)(b * COUT + o)) * HH + h) * WW + wcol] = acc[nt][r] + sBias[o];
      }
    }
  } else {      // layers 1-2: packed bf16 NHWC interior (8 consecutive channels -> uint4)
    const int o0 = mtile * 16 + hf * 8;
    unsigned short* obase = outh + (size_t)b * PSAMP + (size_t)(h + 1) * PROW + o0;
    #pragma unroll
    for (int nt = 0; nt < 4; ++nt) {
      const int wcol = nbase + nt * 16 + lrow;
      uint4 pk; unsigned int* pw = (unsigned int*)&pk;
      #pragma unroll
      for (int p = 0; p < 4; ++p) {
        unsigned int lo = f2bf(acc[nt][2 * p]     + sBias[o0 + 2 * p]);
        unsigned int hi = f2bf(acc[nt][2 * p + 1] + sBias[o0 + 2 * p + 1]);
        pw[p] = lo | (hi << 16);
      }
      *(uint4*)(obase + (size_t)(wcol + 1) * 64) = pk;
    }
  }
}

extern "C" void kernel_launch(void* const* d_in, const int* in_sizes, int n_in,
                              void* d_out, int out_size, void* d_ws, size_t ws_size,
                              hipStream_t stream) {
  const float* x   = (const float*)d_in[0];
  const float* w1  = (const float*)d_in[1];
  const float* w2  = (const float*)d_in[2];
  const float* wt1 = (const float*)d_in[3];
  const float* wt2 = (const float*)d_in[4];
  const float* wt3 = (const float*)d_in[5];
  const float* b1  = (const float*)d_in[6];
  const float* b2  = (const float*)d_in[7];
  const float* b3  = (const float*)d_in[8];
  float* out = (float*)d_out;

  char* ws = (char*)d_ws;
  size_t off = 0;
  auto carve = [&](size_t bytes) -> char* {
    char* p = ws + off;
    off = (off + bytes + 255) & ~(size_t)255;
    return p;
  };
  const size_t padBytes = PSAMP * (size_t)BS * 2;                    // 34.6 MB
  unsigned short* bufA = (unsigned short*)carve(padBytes);           // padded NHWC ping
  unsigned short* bufB = (unsigned short*)carve(padBytes);           // padded NHWC pong
  unsigned short* awbf = (unsigned short*)carve((size_t)3 * BS * COUT * KDIM * 2);
  float* pooled = (float*)carve((size_t)BS * CIN * 4);
  float* coeff  = (float*)carve((size_t)BS * COUT * NE * 4);
  float* abias  = (float*)carve((size_t)3 * BS * COUT * 4);

  hipMemsetAsync(bufA, 0, padBytes, stream);
  hipMemsetAsync(bufB, 0, padBytes, stream);
  hipMemsetAsync(pooled, 0, (size_t)BS * CIN * 4, stream);

  k_cvt<<<BS * HH, 256, 0, stream>>>(x, bufA, pooled);
  k_ctrl<<<1, 256, 0, stream>>>(pooled, w1, w2, b1, b2, b3, coeff, abias);
  k_mixw<<<(3 * BS * COUT * KDIM + 255) / 256, 256, 0, stream>>>(coeff, wt1, wt2, wt3, awbf);

  const size_t lds = (size_t)COUT * KDIM * 2 + COUT * 4;   // ~72.3 KB
  (void)hipFuncSetAttribute((const void*)k_conv,
                            hipFuncAttributeMaxDynamicSharedMemorySize, (int)lds);

  k_conv<<<BS * HH, 256, lds, stream>>>(bufA, awbf, abias, bufB, nullptr);
  k_conv<<<BS * HH, 256, lds, stream>>>(bufB, awbf + (size_t)BS * COUT * KDIM,
                                        abias + BS * COUT, bufA, nullptr);
  k_conv<<<BS * HH, 256, lds, stream>>>(bufA, awbf + (size_t)2 * BS * COUT * KDIM,
                                        abias + 2 * BS * COUT, nullptr, out);
}
